// AttentionPoolingLayer_7816840478840
// MI455X (gfx1250) — matrix operand order repaired and checked by reading, not compiled, not run on hardware
//
#include <hip/hip_runtime.h>

typedef __attribute__((ext_vector_type(16))) _Float16 v16h;
typedef __attribute__((ext_vector_type(8)))  _Float16 v8h;
typedef __attribute__((ext_vector_type(2)))  _Float16 v2h;
typedef __attribute__((ext_vector_type(8)))  float    v8f;
typedef __attribute__((ext_vector_type(2)))  float    f32x2;

#define B_    2048
#define T_    200
#define TPAD_ 208          // 13 * 16
#define D_    64
#define H1_   256
#define H2_   128
#define H3_   64
#define F_    256          // 4*D
#define LDSTRIDE 264       // 256 + 8 halves padding (bank-conflict avoidance)

// ---- workspace layout (units: halves) -------------------------------------
#define W1T_OFF 0
#define W2T_OFF (H1_ * F_)               // 65536
#define W3T_OFF (W2T_OFF + H2_ * H1_)    // 98304
#define WT_TOTAL (W3T_OFF + H3_ * H2_)   // 106496
#define A1T_OFF WT_TOTAL                 // alpha -> [col][TPAD] f16, zero pad
#define A2T_OFF (A1T_OFF + H1_ * TPAD_)
#define A3T_OFF (A2T_OFF + H2_ * TPAD_)
#define B1H_OFF (A3T_OFF + H3_ * TPAD_)  // f16 biases
#define B2H_OFF (B1H_OFF + H1_)
#define B3H_OFF (B2H_OFF + H2_)
#define WS_TOTAL (B3H_OFF + H3_)         // 200128 halves = ~391 KB

// ---------------------------------------------------------------------------
__global__ void convert_params_kernel(const float* __restrict__ W1,
                                      const float* __restrict__ W2,
                                      const float* __restrict__ W3,
                                      const float* __restrict__ a1,
                                      const float* __restrict__ a2,
                                      const float* __restrict__ a3,
                                      const float* __restrict__ b1,
                                      const float* __restrict__ b2,
                                      const float* __restrict__ b3,
                                      _Float16* __restrict__ ws) {
    int idx = blockIdx.x * blockDim.x + threadIdx.x;
    if (idx < W2T_OFF) {                        // W1T
        int n = idx / F_, kk = idx % F_;
        ws[idx] = (_Float16)W1[kk * H1_ + n];
    } else if (idx < W3T_OFF) {                 // W2T
        int j = idx - W2T_OFF, n = j / H1_, kk = j % H1_;
        ws[idx] = (_Float16)W2[kk * H2_ + n];
    } else if (idx < WT_TOTAL) {                // W3T
        int j = idx - W3T_OFF, n = j / H2_, kk = j % H2_;
        ws[idx] = (_Float16)W3[kk * H3_ + n];
    } else if (idx < A2T_OFF) {                 // a1T
        int j = idx - A1T_OFF, n = j / TPAD_, t = j % TPAD_;
        ws[idx] = (t < T_) ? (_Float16)a1[t * H1_ + n] : (_Float16)0.0f;
    } else if (idx < A3T_OFF) {                 // a2T
        int j = idx - A2T_OFF, n = j / TPAD_, t = j % TPAD_;
        ws[idx] = (t < T_) ? (_Float16)a2[t * H2_ + n] : (_Float16)0.0f;
    } else if (idx < B1H_OFF) {                 // a3T
        int j = idx - A3T_OFF, n = j / TPAD_, t = j % TPAD_;
        ws[idx] = (t < T_) ? (_Float16)a3[t * H3_ + n] : (_Float16)0.0f;
    } else if (idx < B2H_OFF) {
        ws[idx] = (_Float16)b1[idx - B1H_OFF];
    } else if (idx < B3H_OFF) {
        ws[idx] = (_Float16)b2[idx - B2H_OFF];
    } else if (idx < WS_TOTAL) {
        ws[idx] = (_Float16)b3[idx - B3H_OFF];
    }
}

__global__ void zero_out_kernel(float* __restrict__ out, int n) {
    int i = blockIdx.x * blockDim.x + threadIdx.x;
    if (i < n) out[i] = 0.0f;
}

// ---------------------------------------------------------------------------
// Packed f16 PReLU epilogue for one 16x16 D tile.
// ---------------------------------------------------------------------------
__device__ __forceinline__ void prelu_store(const v8f& acc, _Float16 bh,
                                            const _Float16* __restrict__ aptr,
                                            _Float16 (*out)[LDSTRIDE],
                                            int rowbase, int col) {
    const v8h al8 = *(const v8h*)aptr;
    v8h hv;
    #pragma unroll
    for (int j = 0; j < 8; ++j) hv[j] = (_Float16)acc[j];
    v8h bb = {bh, bh, bh, bh, bh, bh, bh, bh};
    hv = hv + bb;
    v8h z   = {};
    v8h pos = __builtin_elementwise_max(hv, z);
    v8h neg = __builtin_elementwise_min(hv, z);
    v8h res = pos + al8 * neg;
    #pragma unroll
    for (int j = 0; j < 8; ++j)
        out[rowbase + j][col] = res[j];
}

// ---------------------------------------------------------------------------
// 32-row GEMM stage (two 16-row WMMA tiles sharing every B fragment):
//   out[32][NOUT] = prelu(in[32][KDIM] x Wt + bias)
// ---------------------------------------------------------------------------
template<int KDIM, int NOUT>
__device__ __forceinline__ void gemm2_prelu(const _Float16 (*in)[LDSTRIDE],
                                            _Float16 (*out)[LDSTRIDE],
                                            const _Float16* __restrict__ Wt,
                                            const _Float16* __restrict__ biasH,
                                            const _Float16* __restrict__ alphaT,
                                            int t0, int lane) {
    constexpr int NK = KDIM / 32;
    const int row = lane & 15;
    const int hi  = lane >> 4;

    v16h afA[NK], afB[NK];                         // A fragments, both tiles
    #pragma unroll
    for (int kc = 0; kc < NK; ++kc) {
        const _Float16* ap0 = &in[row][kc * 32 + (hi ? 16 : 0)];
        const _Float16* ap1 = &in[16 + row][kc * 32 + (hi ? 16 : 0)];
        v8h lo0 = *(const v8h*)ap0, hb0 = *(const v8h*)(ap0 + 8);
        v8h lo1 = *(const v8h*)ap1, hb1 = *(const v8h*)(ap1 + 8);
        afA[kc] = __builtin_shufflevector(lo0, hb0, 0,1,2,3,4,5,6,7,8,9,10,11,12,13,14,15);
        afB[kc] = __builtin_shufflevector(lo1, hb1, 0,1,2,3,4,5,6,7,8,9,10,11,12,13,14,15);
    }

    #pragma unroll
    for (int nt = 0; nt < NOUT / 16; ++nt) {
        v8f acc0 = {}, acc1 = {};
        const int col = nt * 16 + row;
        const _Float16* wcol = Wt + (size_t)col * KDIM + (hi ? 16 : 0);
        #pragma unroll
        for (int kc = 0; kc < NK; ++kc) {
            v16h bf = *(const v16h*)(wcol + kc * 32);        // 32B, used twice
            acc0 = __builtin_amdgcn_wmma_f32_16x16x32_f16(
                       false, afA[kc], false, bf, (short)0, acc0, false, false);
            acc1 = __builtin_amdgcn_wmma_f32_16x16x32_f16(
                       false, afB[kc], false, bf, (short)0, acc1, false, false);
        }
        const _Float16 bh = biasH[col];
        const _Float16* abase = alphaT + (size_t)col * TPAD_ + t0 + hi * 8;
        prelu_store(acc0, bh, abase,      out, hi * 8,      col);
        prelu_store(acc1, bh, abase + 16, out, 16 + hi * 8, col);
    }
}

// ---------------------------------------------------------------------------
// Main: one wave32 per (batch b, 32-row group of T); 7 groups cover T=200.
// ---------------------------------------------------------------------------
__global__ __launch_bounds__(32)
void din_attn_kernel(const float* __restrict__ q,  const float* __restrict__ k,
                     const float* __restrict__ Wl, const float* __restrict__ bl,
                     const _Float16* __restrict__ ws, float* __restrict__ out) {
    __shared__ _Float16 bufA[32][LDSTRIDE];
    __shared__ _Float16 bufB[32][LDSTRIDE];
    __shared__ float scores[32];

    const int grp  = blockIdx.x;           // 0..6
    const int b    = blockIdx.y;           // 0..2047
    const int lane = threadIdx.x;          // 0..31
    const int t0   = grp * 32;
    int rmax = T_ - t0;                    // valid rows in this 32-row group
    rmax = rmax < 0 ? 0 : (rmax > 32 ? 32 : rmax);

    // ---- stage 0: att_in = [q | k | q-k | q*k] as f16 rows in LDS ---------
    const int f = 2 * lane;                                 // 2 features/lane
    f32x2 qv = *(const f32x2*)(q + (size_t)b * D_ + f);
    for (int r = 0; r < 32; ++r) {
        float q0 = 0.f, q1 = 0.f, k0 = 0.f, k1 = 0.f;
        if (r < rmax) {
            const float* krow = k + ((size_t)b * T_ + (t0 + r)) * D_;
            f32x2 kv = *(const f32x2*)(krow + f);
            q0 = qv[0]; q1 = qv[1]; k0 = kv[0]; k1 = kv[1];
        }
        *(v2h*)&bufA[r][f]       = v2h{(_Float16)q0,        (_Float16)q1};
        *(v2h*)&bufA[r][64 + f]  = v2h{(_Float16)k0,        (_Float16)k1};
        *(v2h*)&bufA[r][128 + f] = v2h{(_Float16)(q0 - k0), (_Float16)(q1 - k1)};
        *(v2h*)&bufA[r][192 + f] = v2h{(_Float16)(q0 * k0), (_Float16)(q1 * k1)};
    }
    __syncthreads();

    // ---- 3 WMMA GEMM + PReLU stages (M=32) --------------------------------
    gemm2_prelu<F_,  H1_>(bufA, bufB, ws + W1T_OFF, ws + B1H_OFF, ws + A1T_OFF, t0, lane);
    __syncthreads();
    gemm2_prelu<H1_, H2_>(bufB, bufA, ws + W2T_OFF, ws + B2H_OFF, ws + A2T_OFF, t0, lane);
    __syncthreads();
    gemm2_prelu<H2_, H3_>(bufA, bufB, ws + W3T_OFF, ws + B3H_OFF, ws + A3T_OFF, t0, lane);
    __syncthreads();

    // ---- scores: h3[r][:] . Wl + bl, masked on k[b,t,0]; one row per lane --
    {
        float s = 0.0f;
        if (lane < rmax) {
            int t = t0 + lane;
            float acc = bl[0];
            #pragma unroll
            for (int c = 0; c < 8; ++c) {
                v8h hv = *(const v8h*)&bufB[lane][c * 8];   // 16B ds load
                #pragma unroll
                for (int j = 0; j < 8; ++j)
                    acc = fmaf((float)hv[j], Wl[c * 8 + j], acc);
            }
            float kk0 = k[((size_t)b * T_ + t) * D_];
            s = (kk0 != 0.0f) ? acc : 0.0f;
        }
        scores[lane] = s;
    }
    __syncthreads();

    // ---- out[b,d] += sum_t score[t] * k[b,t,d] ----------------------------
    float acc0 = 0.0f, acc1 = 0.0f;
    const int d0 = lane, d1 = lane + 32;
    for (int r = 0; r < rmax; ++r) {
        float sc = scores[r];
        const float* krow = k + ((size_t)b * T_ + (t0 + r)) * D_;
        acc0 = fmaf(sc, krow[d0], acc0);
        acc1 = fmaf(sc, krow[d1], acc1);
    }
    atomicAdd(&out[b * D_ + d0], acc0);
    atomicAdd(&out[b * D_ + d1], acc1);
}

// ---------------------------------------------------------------------------
extern "C" void kernel_launch(void* const* d_in, const int* in_sizes, int n_in,
                              void* d_out, int out_size, void* d_ws, size_t ws_size,
                              hipStream_t stream) {
    (void)in_sizes; (void)n_in; (void)out_size; (void)ws_size;
    const float* q  = (const float*)d_in[0];
    const float* k  = (const float*)d_in[1];
    const float* W1 = (const float*)d_in[2];
    const float* b1 = (const float*)d_in[3];
    const float* a1 = (const float*)d_in[4];
    const float* W2 = (const float*)d_in[5];
    const float* b2 = (const float*)d_in[6];
    const float* a2 = (const float*)d_in[7];
    const float* W3 = (const float*)d_in[8];
    const float* b3 = (const float*)d_in[9];
    const float* a3 = (const float*)d_in[10];
    const float* Wl = (const float*)d_in[11];
    const float* bl = (const float*)d_in[12];
    float*      out = (float*)d_out;
    _Float16*   ws  = (_Float16*)d_ws;

    convert_params_kernel<<<(WS_TOTAL + 255) / 256, 256, 0, stream>>>(
        W1, W2, W3, a1, a2, a3, b1, b2, b3, ws);
    zero_out_kernel<<<(B_ * D_ + 255) / 256, 256, 0, stream>>>(out, B_ * D_);

    dim3 grid(7, B_);    // 7 groups of 32 rows cover T=200; one wave32 each
    din_attn_kernel<<<grid, 32, 0, stream>>>(q, k, Wl, bl, ws, out);
}
